// MRConv2d_plg_14826227105877
// MI455X (gfx1250) — matrix operand order repaired and checked by reading, not compiled, not run on hardware
//
#include <hip/hip_runtime.h>

typedef __attribute__((ext_vector_type(16))) __bf16 v16bf;
typedef __attribute__((ext_vector_type(8)))  float  v8f;

#define Bc   4
#define Cc   384
#define C2   768      // 2*C (GEMM K dimension)
#define Nl   2048     // N_LAB
#define Np   4096     // N_PATCH
#define Kn   16       // neighbors per node
#define NT   32       // n-tile per block (2 WMMA n-subtiles)
#define KSTEPS (C2 / 32)          // 24 k-steps
#define WFRAG_ELEMS (Cc * C2)     // 294912 bf16 elements
#define WFRAG_BYTES (WFRAG_ELEMS * 2)

// Map feats element (row kf in [0,768), col t in [0,32)) to its position in the
// B-fragment-ordered LDS tile: fragment (ks, sub, lane) holds, for lane with
// n = lane&15 and hi = lane>>4, the 16 bf16 values K = ks*32 + hi*8 + (i<8 ? i : i+8).
__device__ __forceinline__ int feat_elem(int kf, int t) {
    int ks   = kf >> 5;
    int hi   = (kf >> 3) & 1;
    int i    = (kf & 7) | (((kf >> 4) & 1) << 3);
    int sub  = t >> 4;
    int lane = (t & 15) | (hi << 4);
    return (((ks * 2 + sub) * 32) + lane) * 16 + i;
}

// Pre-convert W (f32, 384x768 row-major) into bf16 A-fragment order:
// Wfrag[((mtile*24 + ks)*32 + lane)*16 + i] = bf16(W[m, k]),
// m = mtile*16 + (lane&15), k = ks*32 + (lane>>4)*8 + (i<8 ? i : i+8).
__global__ __launch_bounds__(256)
void convert_W(const float* __restrict__ W, __bf16* __restrict__ Wfrag) {
    int e = blockIdx.x * 256 + threadIdx.x;       // 0 .. 294911
    int i    = e & 15;
    int lane = (e >> 4) & 31;
    int ks   = (e >> 9) % KSTEPS;
    int mt   = e / (KSTEPS * 512);
    int m = mt * 16 + (lane & 15);
    int k = ks * 32 + ((lane >> 4) * 8) + (i < 8 ? i : i + 8);
    Wfrag[e] = (__bf16)W[(size_t)m * C2 + k];
}

template <bool USE_WFRAG>
__global__ __launch_bounds__(768)
void mrconv2d_fused(const float*  __restrict__ lab,
                    const float*  __restrict__ patch,
                    const int*    __restrict__ eidx,
                    const float*  __restrict__ Wf,
                    const __bf16* __restrict__ Wfrag,
                    const float*  __restrict__ bias,
                    float*        __restrict__ out)
{
    __shared__ int   s_ii[NT * Kn];                 // lab-side indices  (edge_index[1])
    __shared__ int   s_jj[NT * Kn];                 // patch-side indices (edge_index[0])
    __shared__ v16bf s_fragB[KSTEPS * 2 * 32];      // feats tile in B-fragment order (48 KB)

    const int tid = threadIdx.x;
    const int bid = blockIdx.x;
    const int b   = bid / (Nl / NT);
    const int n0  = (bid % (Nl / NT)) * NT;

    // ---- stage edge indices for this n-tile -------------------------------
    const int* e0 = eidx;                           // x_j (patch) indices
    const int* e1 = eidx + (size_t)Bc * Nl * Kn;    // x_i (lab) indices
    for (int u = tid; u < 2 * NT * Kn; u += 768) {  // 1024 indices
        int which = u >> 9;
        int v = u & 511;
        int t = v >> 4, k = v & 15;
        int val = (which ? e0 : e1)[((size_t)b * Nl + n0 + t) * Kn + k];
        if (which) s_jj[v] = val; else s_ii[v] = val;
    }
    __syncthreads();

    const float* labB = lab   + (size_t)b * Cc * Nl;
    const float* patB = patch + (size_t)b * Cc * Np;
    __bf16* fragw = (__bf16*)s_fragB;

    // ---- phase 1: build feats tile (768 x 32) directly in fragment order --
    // rows 0..383: bf16(lab[c, n0+t])       (coalesced: one c per 32-lane group)
    #pragma unroll 4
    for (int it = 0; it < (Cc * NT) / 768; ++it) {  // 16 iters
        int idx = it * 768 + tid;
        int c = idx >> 5;
        int t = idx & 31;
        fragw[feat_elem(c, t)] = (__bf16)labB[(size_t)c * Nl + n0 + t];
    }
    // rows 384..767: bf16(max_k(lab[c, ii] - patch[c, jj]))  (reduction in f32)
    #pragma unroll 2
    for (int it = 0; it < (Cc * NT) / 768; ++it) {  // 16 iters
        int idx = it * 768 + tid;
        int c = idx >> 5;
        int t = idx & 31;
        const float* lrow = labB + (size_t)c * Nl;
        const float* prow = patB + (size_t)c * Np;
        float m = -3.0e38f;
        #pragma unroll
        for (int k = 0; k < Kn; ++k) {
            float d = lrow[s_ii[t * Kn + k]] - prow[s_jj[t * Kn + k]];
            m = fmaxf(m, d);
        }
        fragw[feat_elem(Cc + c, t)] = (__bf16)m;
    }
    __syncthreads();

    // ---- phase 2: WMMA GEMM, each wave: 16 rows x 32 cols -----------------
    const int lane = tid & 31;
    const int wave = tid >> 5;              // 0..23 -> M-tile
    const int o0   = wave * 16;

    v8f acc0 = {0.f,0.f,0.f,0.f,0.f,0.f,0.f,0.f};
    v8f acc1 = {0.f,0.f,0.f,0.f,0.f,0.f,0.f,0.f};

    #pragma unroll 4
    for (int ks = 0; ks < KSTEPS; ++ks) {
        // A fragment
        v16bf a;
        if (USE_WFRAG) {
            // one contiguous 32B load per lane; wave reads 1KB contiguous
            a = *(const v16bf*)(Wfrag + (((size_t)wave * KSTEPS + ks) * 32 + lane) * 16);
        } else {
            const int mrow = o0 + (lane & 15);
            const int kA   = (lane >> 4) * 8;
            const float4* wr = (const float4*)(Wf + (size_t)mrow * C2 + ks * 32 + kA);
            float4 wa = wr[0], wb = wr[1], wc = wr[4], wd = wr[5];
            a[0]=(__bf16)wa.x;  a[1]=(__bf16)wa.y;  a[2]=(__bf16)wa.z;  a[3]=(__bf16)wa.w;
            a[4]=(__bf16)wb.x;  a[5]=(__bf16)wb.y;  a[6]=(__bf16)wb.z;  a[7]=(__bf16)wb.w;
            a[8]=(__bf16)wc.x;  a[9]=(__bf16)wc.y;  a[10]=(__bf16)wc.z; a[11]=(__bf16)wc.w;
            a[12]=(__bf16)wd.x; a[13]=(__bf16)wd.y; a[14]=(__bf16)wd.z; a[15]=(__bf16)wd.w;
        }
        // B fragments: contiguous 32B per lane from LDS (2x ds_load_b128 each)
        v16bf b0 = s_fragB[(ks * 2 + 0) * 32 + lane];
        v16bf b1 = s_fragB[(ks * 2 + 1) * 32 + lane];

        acc0 = __builtin_amdgcn_wmma_f32_16x16x32_bf16(false, a, false, b0,
                                                       (short)0, acc0, false, false);
        acc1 = __builtin_amdgcn_wmma_f32_16x16x32_bf16(false, a, false, b1,
                                                       (short)0, acc1, false, false);
    }

    // ---- store with bias + ReLU -------------------------------------------
    // D layout: VGPR r -> M = r (lanes 0-15) or r+8 (lanes 16-31), N = lane&15
    const int mo = o0 + ((lane >> 4) * 8);
    const int nA = n0 + (lane & 15);
    const int nBcol = nA + 16;
    float* outB = out + (size_t)b * Cc * Nl;
    #pragma unroll
    for (int r = 0; r < 8; ++r) {
        int o = mo + r;
        float bo = bias[o];
        outB[(size_t)o * Nl + nA]    = fmaxf(acc0[r] + bo, 0.f);
        outB[(size_t)o * Nl + nBcol] = fmaxf(acc1[r] + bo, 0.f);
    }
}

extern "C" void kernel_launch(void* const* d_in, const int* in_sizes, int n_in,
                              void* d_out, int out_size, void* d_ws, size_t ws_size,
                              hipStream_t stream) {
    (void)in_sizes; (void)n_in; (void)out_size;
    const float* lab   = (const float*)d_in[0];
    const float* patch = (const float*)d_in[1];
    const int*   eidx  = (const int*)d_in[2];
    const float* W     = (const float*)d_in[3];
    const float* bias  = (const float*)d_in[4];
    float* out = (float*)d_out;

    dim3 grid(Bc * (Nl / NT));   // 256 blocks: one per (b, 32-column n-tile)
    dim3 block(768);             // 24 waves: one 16-row M-tile each

    if (ws_size >= (size_t)WFRAG_BYTES) {
        __bf16* Wfrag = (__bf16*)d_ws;
        hipLaunchKernelGGL(convert_W, dim3(WFRAG_ELEMS / 256), dim3(256), 0, stream,
                           W, Wfrag);
        hipLaunchKernelGGL((mrconv2d_fused<true>), grid, block, 0, stream,
                           lab, patch, eidx, W, Wfrag, bias, out);
    } else {
        hipLaunchKernelGGL((mrconv2d_fused<false>), grid, block, 0, stream,
                           lab, patch, eidx, W, (const __bf16*)nullptr, bias, out);
    }
}
